// Model_51926154609432
// MI455X (gfx1250) — matrix operand order repaired and checked by reading, not compiled, not run on hardware
//
#include <hip/hip_runtime.h>
#include <hip/hip_bf16.h>
#include <math.h>

typedef __attribute__((ext_vector_type(16))) _Float16 v16h;
typedef __attribute__((ext_vector_type(8)))  float    v8f;

#define NB   104          // B*CP = 8*13
#define NN   207
#define DD   64
#define ELEMS ((size_t)NB * NN * DD)

__device__ __forceinline__ float sigmoidf_(float x) { return 1.f / (1.f + __expf(-x)); }

// ---------------------------------------------------------------------------
// Dynamic parameter generation (4 weight sets + biases).  One block per
// (i, b, c); tiny MLP chains, pure VALU.
// Wp: (4, 104, 64, 64)  Bp: (4, 104, 64)
// ---------------------------------------------------------------------------
__global__ void __launch_bounds__(64)
param_gen(const float* __restrict__ h_time, const float* __restrict__ h_space,
          const float* __restrict__ Ttw,    const float* __restrict__ Ttb,
          const float* __restrict__ wg_w1,  const float* __restrict__ wg_b1,
          const float* __restrict__ wg_w2,  const float* __restrict__ wg_b2,
          const float* __restrict__ wg_w3,  const float* __restrict__ wg_b3,
          const float* __restrict__ bg_w1,  const float* __restrict__ bg_b1,
          const float* __restrict__ bg_w2,  const float* __restrict__ bg_b2,
          const float* __restrict__ bg_w3,  const float* __restrict__ bg_b3,
          float* __restrict__ Wp, float* __restrict__ Bp)
{
    const int blk = blockIdx.x;          // 4 * 104
    const int i   = blk / NB;
    const int bc  = blk % NB;
    const int b   = bc / 13;
    const int c   = bc % 13;
    const float* hs = (i < 2) ? h_time : h_space;

    __shared__ float mem[64], m1w[32], m1b[32], m2w[5], m2b[5];
    const int tid = threadIdx.x;

    if (tid < 64) {
        float s = 0.f;
        for (int l = 0; l < 48; ++l)
            s += tanhf(hs[((size_t)b * 48 + l) * 64 + tid]) * Ttw[((size_t)i * 13 + c) * 48 + l];
        mem[tid] = tanhf(s + Ttb[i * 13 + c]);
    }
    __syncthreads();
    if (tid < 32) {
        float s1 = wg_b1[i * 32 + tid], s2 = bg_b1[i * 32 + tid];
        for (int h = 0; h < 64; ++h) {
            const float m = mem[h];
            s1 += m * wg_w1[((size_t)i * 32 + tid) * 64 + h];
            s2 += m * bg_w1[((size_t)i * 32 + tid) * 64 + h];
        }
        m1w[tid] = fmaxf(s1, 0.f);
        m1b[tid] = fmaxf(s2, 0.f);
    }
    __syncthreads();
    if (tid < 5) {
        float s1 = wg_b2[i * 5 + tid], s2 = bg_b2[i * 5 + tid];
        for (int o = 0; o < 32; ++o) {
            s1 += m1w[o] * wg_w2[((size_t)i * 5 + tid) * 32 + o];
            s2 += m1b[o] * bg_w2[((size_t)i * 5 + tid) * 32 + o];
        }
        m2w[tid] = fmaxf(s1, 0.f);
        m2b[tid] = fmaxf(s2, 0.f);
    }
    __syncthreads();

    const float t0 = m2w[0], t1 = m2w[1], t2 = m2w[2], t3 = m2w[3], t4 = m2w[4];
    for (int f = tid; f < 4096; f += 64) {
        const float* wr = wg_w3 + ((size_t)i * 4096 + f) * 5;
        float v = wg_b3[(size_t)i * 4096 + f]
                + t0 * wr[0] + t1 * wr[1] + t2 * wr[2] + t3 * wr[3] + t4 * wr[4];
        Wp[((size_t)i * NB + bc) * 4096 + f] = v;
    }
    if (tid < 64) {
        const float u0 = m2b[0], u1 = m2b[1], u2 = m2b[2], u3 = m2b[3], u4 = m2b[4];
        const float* br = bg_w3 + ((size_t)i * 64 + tid) * 5;
        float v = bg_b3[(size_t)i * 64 + tid]
                + u0 * br[0] + u1 * br[1] + u2 * br[2] + u3 * br[3] + u4 * br[4];
        Bp[((size_t)i * NB + bc) * 64 + tid] = v;
    }
}

// ---------------------------------------------------------------------------
// Batched WMMA GEMM: Y[bt] = act( X[bt](207x64) @ B(64x64) + bias ).
// per_batch=1: B = W[bt] (lin_custom, direct d,e);  per_batch=0: B = W^T (dense).
// Block = 4 waves; wave w computes one 16x16 output tile (cols 16w..16w+15);
// K=64 split into 2 x v_wmma_f32_16x16x32_f16.  act: 0=none, 1=relu, 2=tanh.
// A staged row-major, B staged COLUMN-major (BhT[e][k]) so every fragment is
// contiguous 16/32B runs in LDS -> ds_load_b128 instead of 16x ds_load_u16.
// ---------------------------------------------------------------------------
__global__ void __launch_bounds__(128)
gemm_bt64(const float* __restrict__ X, const float* __restrict__ W,
          const float* __restrict__ Bv, float* __restrict__ Y,
          int per_batch, int transposedW, int act)
{
    const int bt   = blockIdx.y;
    const int row0 = blockIdx.x * 16;
    const int tid  = threadIdx.x;
    const int lane = tid & 31;
    const int wv   = tid >> 5;
    const int nn   = lane & 15;
    const int gg   = lane >> 4;

    __shared__ alignas(16) _Float16 Ah[16][72];     // [m][k]
    __shared__ alignas(16) _Float16 BhT[64][72];    // [e][k]

    const float* Xb = X + (size_t)bt * NN * DD;
    const float* Wb = per_batch ? (W + (size_t)bt * 64 * 64) : W;
    const float* bb = per_batch ? (Bv + (size_t)bt * 64) : Bv;

    __builtin_prefetch(Xb + (size_t)row0 * 64, 0, 1);
    __builtin_prefetch(Wb, 0, 1);

    {   // stage A tile (16 x 64) -> f16
        const int r  = tid >> 3;
        const int c0 = (tid & 7) * 8;
        const int gr = row0 + r;
        if (gr < NN) {
            #pragma unroll
            for (int j = 0; j < 8; ++j) Ah[r][c0 + j] = (_Float16)Xb[(size_t)gr * 64 + c0 + j];
        } else {
            #pragma unroll
            for (int j = 0; j < 8; ++j) Ah[r][c0 + j] = (_Float16)0.f;
        }
    }
    if (transposedW) {          // dense: BhT[e][k] = W[e*64+k]  (coalesced both sides)
        for (int idx = tid; idx < 4096; idx += 128) {
            const int e = idx >> 6, k = idx & 63;
            BhT[e][k] = (_Float16)Wb[(size_t)e * 64 + k];
        }
    } else {                    // lin_custom: BhT[e][k] = W[k*64+e]
        for (int idx = tid; idx < 4096; idx += 128) {
            const int k = idx >> 6, e = idx & 63;
            BhT[e][k] = (_Float16)Wb[idx];
        }
    }
    __syncthreads();

    const int col = wv * 16;
    v8f c = {};
    #pragma unroll
    for (int ks = 0; ks < 2; ++ks) {
        v16h a, bf;
        #pragma unroll
        for (int j = 0; j < 16; ++j) {   // A frag: m=nn, k = g*16 + j (contiguous)
            const int kl = ks * 32 + gg * 16 + j;
            a[j] = Ah[nn][kl];
        }
        #pragma unroll
        for (int j = 0; j < 16; ++j) {   // B frag: n=nn, k = g*16 + j (contiguous)
            bf[j] = BhT[col + nn][ks * 32 + gg * 16 + j];
        }
        c = __builtin_amdgcn_wmma_f32_16x16x32_f16(false, a, false, bf, (short)0, c, false, false);
    }

    const float bias = bb[col + nn];
    #pragma unroll
    for (int r = 0; r < 8; ++r) {        // C frag: m = g*8 + r, n = nn
        const int gr = row0 + gg * 8 + r;
        if (gr < NN) {
            float v = c[r] + bias;
            if (act == 1)      v = fmaxf(v, 0.f);
            else if (act == 2) v = tanhf(v);
            Y[((size_t)bt * NN + gr) * 64 + col + nn] = v;
        }
    }
}

// ---------------------------------------------------------------------------
// Temporal attention (13x13 per (b,n), 8 heads) + softmax + att@V.  Tiny ->
// VALU.  Writes merged-head xm (8,13,207,64).
// ---------------------------------------------------------------------------
__global__ void __launch_bounds__(128)
temporal_attn(const float* __restrict__ T1, const float* __restrict__ Kb,
              const float* __restrict__ Vb, const float* __restrict__ rate,
              float* __restrict__ XM)
{
    const int bn = blockIdx.x;
    const int b  = bn / NN;
    const int n  = bn % NN;
    const int tid = threadIdx.x;

    __shared__ alignas(16) float q[64];
    __shared__ alignas(16) float kk[13][64];
    __shared__ alignas(16) float vv[13][64];
    __shared__ float sa[8][13];

    if (tid < 64) q[tid] = T1[(((size_t)b * 13) * NN + n) * 64 + tid];
    for (int idx = tid; idx < 13 * 64; idx += 128) {
        const int t = idx >> 6, d = idx & 63;
        const size_t off = (((size_t)b * 13 + t) * NN + n) * 64 + d;
        kk[t][d] = Kb[off];
        vv[t][d] = Vb[off];
    }
    __syncthreads();

    const float inv = 0.35355339059f;    // 1/sqrt(8)
    if (tid < 104) {
        const int h = tid / 13, j = tid % 13;
        float s = 0.f;
        #pragma unroll
        for (int e = 0; e < 8; ++e) s += q[h * 8 + e] * kk[j][h * 8 + e];
        sa[h][j] = sigmoidf_(s * inv);
    }
    __syncthreads();

    if (tid < 104) {
        const float srate = sigmoidf_(rate[0]);
        const int h = tid / 13, i = tid % 13;
        const float dec = (float)(13 - i);
        float row[13];
        float mx = -1e30f;
        #pragma unroll
        for (int j = 0; j < 13; ++j) {
            const float s = sa[h][j];
            const float g = sigmoidf_(s * srate * dec);
            const float a = tanhf(s / (1.f + g));
            row[j] = a;
            mx = fmaxf(mx, a);
        }
        float sum = 0.f;
        #pragma unroll
        for (int j = 0; j < 13; ++j) { row[j] = __expf(row[j] - mx); sum += row[j]; }
        const float rinv = 1.f / sum;
        #pragma unroll
        for (int e = 0; e < 8; ++e) {
            float acc = 0.f;
            #pragma unroll
            for (int j = 0; j < 13; ++j) acc += row[j] * vv[j][h * 8 + e];
            XM[(((size_t)b * 13 + i) * NN + n) * 64 + h * 8 + e] = acc * rinv;
        }
    }
}

// ---------------------------------------------------------------------------
// Fused flash-style spatial attention.  One block per (kb*13+t, qtile).
// Q/K staged feature-padded to 32 (zeros live in LDS, so fragments need no
// per-element masking); V staged TRANSPOSED (VhT[e][n], rows 8..15 zero).
// All WMMA fragments read contiguous 16/32B LDS runs -> ds_load_b128.
// The 207x207 score matrix never touches HBM.
// ---------------------------------------------------------------------------
__global__ void __launch_bounds__(256)
spatial_attn(const float* __restrict__ Q, const float* __restrict__ Ks,
             const float* __restrict__ Vs, float* __restrict__ O)
{
    const int qtile = blockIdx.x;            // 0..12
    const int kbT   = blockIdx.y;            // 0..831
    const int kb    = kbT / 13;
    const int t     = kbT % 13;
    const int head  = kb >> 3;
    const int b     = kb & 7;
    const int bt    = b * 13 + t;
    const int dbase = head * 8;
    const int qrow0 = qtile * 16;

    const int tid  = threadIdx.x;
    const int lane = tid & 31;
    const int wv   = tid >> 5;               // 8 waves
    const int nn   = lane & 15;
    const int gg   = lane >> 4;

    __shared__ alignas(16) _Float16 Qh[16][32];     // [q][k], k 8..31 zero
    __shared__ alignas(16) _Float16 Kh[224][32];    // [key][k], k 8..31 zero
    __shared__ alignas(16) _Float16 VhT[16][232];   // [e][key], e 8..15 zero
    __shared__ alignas(16) float    Sb[16][224];
    __shared__ alignas(16) _Float16 Ph[16][232];    // [q][key]
    __shared__ alignas(16) float    part[7][16][8];
    __shared__ float    red[16][16];
    __shared__ float    rowsum[16];
    __shared__ float    rowmax[16];

    for (int idx = tid; idx < 224 * 32; idx += 256) {
        const int n = idx >> 5, e = idx & 31;
        float v = 0.f;
        if (n < NN && e < 8) v = Ks[((size_t)bt * NN + n) * 64 + dbase + e];
        Kh[n][e] = (_Float16)v;
    }
    for (int idx = tid; idx < 16 * 232; idx += 256) {
        const int e = idx / 232, n = idx % 232;
        float v = 0.f;
        if (e < 8 && n < NN) v = Vs[((size_t)bt * NN + n) * 64 + dbase + e];
        VhT[e][n] = (_Float16)v;
    }
    for (int idx = tid; idx < 16 * 32; idx += 256) {
        const int r = idx >> 5, e = idx & 31;
        const int gr = qrow0 + r;
        float v = 0.f;
        if (e < 8 && gr < NN) v = Q[((size_t)bt * NN + gr) * 64 + dbase + e];
        Qh[r][e] = (_Float16)v;
    }
    __syncthreads();

    // ---- scores: S = Q(16x8) @ K^T, one WMMA per 16-key tile ----
    const float scale = 0.35355339059f;
    v16h aq;
    #pragma unroll
    for (int j = 0; j < 16; ++j) {
        aq[j] = Qh[nn][gg * 16 + j];
    }
    for (int kt = wv; kt < 13; kt += 8) {
        v16h bk;
        #pragma unroll
        for (int j = 0; j < 16; ++j) {
            bk[j] = Kh[kt * 16 + nn][gg * 16 + j];
        }
        v8f s = {};
        s = __builtin_amdgcn_wmma_f32_16x16x32_f16(false, aq, false, bk, (short)0, s, false, false);
        #pragma unroll
        for (int r = 0; r < 8; ++r) Sb[gg * 8 + r][kt * 16 + nn] = s[r] * scale;
    }
    __syncthreads();

    for (int idx = tid; idx < 16 * 17; idx += 256) {   // mask padded key columns
        Sb[idx / 17][207 + (idx % 17)] = -1e30f;
    }
    __syncthreads();

    // ---- row softmax (16 rows x 16 threads each) ----
    const int r  = tid >> 4;
    const int ct = tid & 15;
    float mx = -1e30f;
    for (int cc = ct; cc < NN; cc += 16) mx = fmaxf(mx, Sb[r][cc]);
    red[r][ct] = mx;
    __syncthreads();
    if (ct == 0) {
        float m2 = red[r][0];
        #pragma unroll
        for (int q2 = 1; q2 < 16; ++q2) m2 = fmaxf(m2, red[r][q2]);
        rowmax[r] = m2;
    }
    __syncthreads();
    const float mrow = rowmax[r];
    float sm = 0.f;
    for (int cc = ct; cc < 224; cc += 16) {
        const float p = (cc < NN) ? __expf(Sb[r][cc] - mrow) : 0.f;
        Ph[r][cc] = (_Float16)p;
        sm += p;
    }
    red[r][ct] = sm;
    __syncthreads();
    if (ct == 0) {
        float s2 = 0.f;
        #pragma unroll
        for (int q2 = 0; q2 < 16; ++q2) s2 += red[r][q2];
        rowsum[r] = s2;
    }
    __syncthreads();

    // ---- P @ V: 7 K-chunks of 32 keys, one per wave, then reduce ----
    if (wv < 7) {
        v16h ap, bv;
        #pragma unroll
        for (int j = 0; j < 16; ++j) {
            ap[j] = Ph[nn][wv * 32 + gg * 16 + j];
        }
        #pragma unroll
        for (int j = 0; j < 16; ++j) {
            bv[j] = VhT[nn][wv * 32 + gg * 16 + j];
        }
        v8f c = {};
        c = __builtin_amdgcn_wmma_f32_16x16x32_f16(false, ap, false, bv, (short)0, c, false, false);
        #pragma unroll
        for (int r2 = 0; r2 < 8; ++r2) {
            if (nn < 8) part[wv][gg * 8 + r2][nn] = c[r2];
        }
    }
    __syncthreads();
    if (tid < 128) {
        const int m = tid >> 3, e = tid & 7;
        float acc = 0.f;
        #pragma unroll
        for (int w = 0; w < 7; ++w) acc += part[w][m][e];
        const int gr = qrow0 + m;
        if (gr < NN) O[((size_t)bt * NN + gr) * 64 + dbase + e] = acc / rowsum[m];
    }
}

// ---------------------------------------------------------------------------
// Build t1 for cut i: row 0 = proxies[cut] + out[cut-1]; rows 1..12 = x slice.
// ---------------------------------------------------------------------------
__global__ void __launch_bounds__(256)
build_t1(const float* __restrict__ x, const float* __restrict__ proxies,
         const float* __restrict__ outPrev, float* __restrict__ T1, int cut)
{
    const size_t idx = (size_t)blockIdx.x * 256 + threadIdx.x;
    if (idx >= ELEMS) return;
    const int d = (int)(idx & 63);
    size_t rest = idx >> 6;
    const int n = (int)(rest % NN); rest /= NN;
    const int t = (int)(rest % 13);
    const int b = (int)(rest / 13);
    float v;
    if (t == 0) {
        v = proxies[((size_t)cut * NN + n) * 64 + d];
        if (cut > 0) v += outPrev[(((size_t)b * 4 + (cut - 1)) * NN + n) * 64 + d];
    } else {
        v = x[(((size_t)b * 48 + cut * 12 + (t - 1)) * NN + n) * 64 + d];
    }
    T1[idx] = v;
}

// ---------------------------------------------------------------------------
// out[cut] = sum_t sigmoid(logits) * o2
// ---------------------------------------------------------------------------
__global__ void __launch_bounds__(256)
gate_reduce(const float* __restrict__ logits, const float* __restrict__ o2,
            float* __restrict__ out, int cut)
{
    const size_t idx = (size_t)blockIdx.x * 256 + threadIdx.x;
    if (idx >= (size_t)8 * NN * DD) return;
    const int d = (int)(idx & 63);
    size_t rest = idx >> 6;
    const int n = (int)(rest % NN);
    const int b = (int)(rest / NN);
    float acc = 0.f;
    for (int t = 0; t < 13; ++t) {
        const size_t off = (((size_t)b * 13 + t) * NN + n) * 64 + d;
        acc += sigmoidf_(logits[off]) * o2[off];
    }
    out[(((size_t)b * 4 + cut) * NN + n) * 64 + d] = acc;
}

// ---------------------------------------------------------------------------
extern "C" void kernel_launch(void* const* d_in, const int* in_sizes, int n_in,
                              void* d_out, int out_size, void* d_ws, size_t ws_size,
                              hipStream_t stream)
{
    (void)in_sizes; (void)n_in; (void)out_size; (void)ws_size;

    const float* x        = (const float*)d_in[0];
    const float* h_time   = (const float*)d_in[1];
    const float* h_space  = (const float*)d_in[2];
    const float* gTtw     = (const float*)d_in[3];
    const float* gTtb     = (const float*)d_in[4];
    const float* wg_w1    = (const float*)d_in[5];
    const float* wg_b1    = (const float*)d_in[6];
    const float* wg_w2    = (const float*)d_in[7];
    const float* wg_b2    = (const float*)d_in[8];
    const float* wg_w3    = (const float*)d_in[9];
    const float* wg_b3    = (const float*)d_in[10];
    const float* bg_w1    = (const float*)d_in[11];
    const float* bg_b1    = (const float*)d_in[12];
    const float* bg_w2    = (const float*)d_in[13];
    const float* bg_b2    = (const float*)d_in[14];
    const float* bg_w3    = (const float*)d_in[15];
    const float* bg_b3    = (const float*)d_in[16];
    const float* proxies  = (const float*)d_in[17];
    const float* ta_p1_w  = (const float*)d_in[18];
    const float* ta_p1_b  = (const float*)d_in[19];
    const float* ta_p2_w  = (const float*)d_in[20];
    const float* ta_p2_b  = (const float*)d_in[21];
    const float* rate     = (const float*)d_in[22];
    const float* sa_p1_w  = (const float*)d_in[23];
    const float* sa_p1_b  = (const float*)d_in[24];
    const float* sa_p2_w  = (const float*)d_in[25];
    const float* sa_p2_b  = (const float*)d_in[26];
    const float* agg_w1   = (const float*)d_in[27];
    const float* agg_b1   = (const float*)d_in[28];
    const float* agg_w2   = (const float*)d_in[29];
    const float* agg_b2   = (const float*)d_in[30];

    float* ws = (float*)d_ws;
    const size_t WSTRIDE = (size_t)NB * 4096;     // one dynamic-weight set
    const size_t BSTRIDE = (size_t)NB * 64;
    float* Wp  = ws;                               // 4 * 104 * 4096
    float* Bp  = Wp + 4 * WSTRIDE;                 // 4 * 104 * 64
    float* b0  = Bp + 4 * BSTRIDE;
    float* b1  = b0 + ELEMS;
    float* b2  = b1 + ELEMS;
    float* b3  = b2 + ELEMS;
    float* b4  = b3 + ELEMS;
    float* b5  = b4 + ELEMS;
    float* out = (float*)d_out;

    param_gen<<<4 * NB, 64, 0, stream>>>(h_time, h_space, gTtw, gTtb,
                                         wg_w1, wg_b1, wg_w2, wg_b2, wg_w3, wg_b3,
                                         bg_w1, bg_b1, bg_w2, bg_b2, bg_w3, bg_b3,
                                         Wp, Bp);

    const dim3 gGemm(13, NB);
    const int  nT1  = (int)((ELEMS + 255) / 256);
    const int  nRed = (int)(((size_t)8 * NN * DD + 255) / 256);

    for (int cut = 0; cut < 4; ++cut) {
        build_t1<<<nT1, 256, 0, stream>>>(x, proxies, out, b0, cut);

        // temporal k/v projections (dynamic per-(b,t) weights, WMMA)
        gemm_bt64<<<gGemm, 128, 0, stream>>>(b0, Wp + 0 * WSTRIDE, Bp + 0 * BSTRIDE, b1, 1, 0, 0);
        gemm_bt64<<<gGemm, 128, 0, stream>>>(b0, Wp + 1 * WSTRIDE, Bp + 1 * BSTRIDE, b2, 1, 0, 0);

        temporal_attn<<<8 * NN, 128, 0, stream>>>(b0, b1, b2, rate, b3);

        gemm_bt64<<<gGemm, 128, 0, stream>>>(b3, ta_p1_w, ta_p1_b, b4, 0, 1, 2);   // tanh
        gemm_bt64<<<gGemm, 128, 0, stream>>>(b4, ta_p2_w, ta_p2_b, b0, 0, 1, 0);   // o1

        // spatial k/v projections (dynamic weights, WMMA)
        gemm_bt64<<<gGemm, 128, 0, stream>>>(b0, Wp + 2 * WSTRIDE, Bp + 2 * BSTRIDE, b1, 1, 0, 0);
        gemm_bt64<<<gGemm, 128, 0, stream>>>(b0, Wp + 3 * WSTRIDE, Bp + 3 * BSTRIDE, b2, 1, 0, 0);

        spatial_attn<<<dim3(13, 832), 256, 0, stream>>>(b0, b1, b2, b3);

        gemm_bt64<<<gGemm, 128, 0, stream>>>(b3, sa_p1_w, sa_p1_b, b4, 0, 1, 1);   // relu
        gemm_bt64<<<gGemm, 128, 0, stream>>>(b4, sa_p2_w, sa_p2_b, b5, 0, 1, 0);   // o2
        gemm_bt64<<<gGemm, 128, 0, stream>>>(b5, agg_w1,  agg_b1,  b4, 0, 1, 1);   // relu
        gemm_bt64<<<gGemm, 128, 0, stream>>>(b4, agg_w2,  agg_b2,  b3, 0, 1, 0);   // gate logits

        gate_reduce<<<nRed, 256, 0, stream>>>(b3, b5, out, cut);
    }
}
